// MultiHeadAttention_42013370089646
// MI455X (gfx1250) — matrix operand order repaired and checked by reading, not compiled, not run on hardware
//
#include <hip/hip_runtime.h>

typedef __attribute__((ext_vector_type(16))) _Float16 v16h;
typedef __attribute__((ext_vector_type(8)))  float    v8f;
typedef _Float16 half_t;

namespace {
constexpr int Bc = 2, Sc = 2048, Dc = 1024, Ec = 1024, Hc = 16, DKc = 64;
}

// ---------------------------------------------------------------------------
// Kernel 1: fused QKV projection.
//   C[4096 x 3072] = X[4096 x 1024] * W_qkv^T, + bias, scatter to per-head
//   f16 Q/K/V buffers laid out [B,H,S,DK].
// Block: 256 threads (8 waves). Tile 128x128, BK=32. Each wave: 32x64.
// Global->VGPR prefetch of the next K-slice overlaps the WMMA work.
// ---------------------------------------------------------------------------
__global__ __launch_bounds__(256) void qkv_gemm_wmma(
    const float* __restrict__ X,      // [B*S, D] fp32
    const float* __restrict__ W,      // [3E, D]  fp32 (torch [out,in])
    const float* __restrict__ bias,   // [3E]
    half_t* __restrict__ Qh, half_t* __restrict__ Kh, half_t* __restrict__ Vh)
{
    __shared__ __align__(16) half_t Xs[128][40];
    __shared__ __align__(16) half_t Ws[128][40];

    const int tid  = threadIdx.x;
    const int lane = tid & 31;
    const int wid  = tid >> 5;
    const int wm   = wid & 3;        // 32-row band
    const int wn   = wid >> 2;       // 64-col band
    const int m0   = blockIdx.y * 128;
    const int n0   = blockIdx.x * 128;
    const int lr   = lane & 15;
    const int hs   = lane >> 4;

    v8f acc[2][4];
#pragma unroll
    for (int mi = 0; mi < 2; ++mi)
#pragma unroll
        for (int ni = 0; ni < 4; ++ni)
#pragma unroll
            for (int r = 0; r < 8; ++r) acc[mi][ni][r] = 0.0f;

    const int lrow = tid >> 3;        // 0..31
    const int lseg = (tid & 7) * 4;   // 0..28 (floats)

    float4 rx[4], rw[4];
    auto loadXW = [&](int k0) {
#pragma unroll
        for (int p = 0; p < 4; ++p) {
            const int row = p * 32 + lrow;
            rx[p] = *(const float4*)(X + (size_t)(m0 + row) * Dc + k0 + lseg);
            rw[p] = *(const float4*)(W + (size_t)(n0 + row) * Dc + k0 + lseg);
        }
    };
    loadXW(0);

    for (int k0 = 0; k0 < Dc; k0 += 32) {
        __syncthreads();
#pragma unroll
        for (int p = 0; p < 4; ++p) {
            const int row = p * 32 + lrow;
            Xs[row][lseg + 0] = (half_t)rx[p].x; Xs[row][lseg + 1] = (half_t)rx[p].y;
            Xs[row][lseg + 2] = (half_t)rx[p].z; Xs[row][lseg + 3] = (half_t)rx[p].w;
            Ws[row][lseg + 0] = (half_t)rw[p].x; Ws[row][lseg + 1] = (half_t)rw[p].y;
            Ws[row][lseg + 2] = (half_t)rw[p].z; Ws[row][lseg + 3] = (half_t)rw[p].w;
        }
        __syncthreads();
        if (k0 + 32 < Dc) loadXW(k0 + 32);   // overlap with WMMAs below

        // A fragments (16x32 f16): lane half selects K {0-7,16-23} / {8-15,24-31}
        v16h afr[2];
#pragma unroll
        for (int mi = 0; mi < 2; ++mi) {
            const int m  = wm * 32 + mi * 16 + lr;
            const int kb = hs * 8;
#pragma unroll
            for (int j = 0; j < 8; ++j) afr[mi][j]     = Xs[m][kb + j];
#pragma unroll
            for (int j = 0; j < 8; ++j) afr[mi][8 + j] = Xs[m][16 + kb + j];
        }
        // B fragments (32x16): lane half selects K {0-7,16-23} / {8-15,24-31}; B[k][n]=W[n][k]
        v16h bfr[4];
#pragma unroll
        for (int ni = 0; ni < 4; ++ni) {
            const int n  = wn * 64 + ni * 16 + lr;
            const int kb = hs * 8;
#pragma unroll
            for (int j = 0; j < 16; ++j) bfr[ni][j] = Ws[n][kb + (j & 7) + ((j >> 3) << 4)];
        }
#pragma unroll
        for (int mi = 0; mi < 2; ++mi)
#pragma unroll
            for (int ni = 0; ni < 4; ++ni)
                acc[mi][ni] = __builtin_amdgcn_wmma_f32_16x16x32_f16(
                    false, afr[mi], false, bfr[ni], (short)0, acc[mi][ni], false, false);
    }

    // Epilogue: column n -> (head, q/k/v, dk); rows -> (b, s)
    const int mb = hs * 8;
#pragma unroll
    for (int mi = 0; mi < 2; ++mi)
#pragma unroll
        for (int ni = 0; ni < 4; ++ni) {
            const int n     = n0 + wn * 64 + ni * 16 + lr;
            const int h     = n / (3 * DKc);
            const int rem   = n - h * 3 * DKc;
            const int which = rem / DKc;
            const int dk    = rem - which * DKc;
            half_t* dst = (which == 0) ? Qh : (which == 1) ? Kh : Vh;
            const float bv = bias[n];
#pragma unroll
            for (int r = 0; r < 8; ++r) {
                const int m = m0 + wm * 32 + mi * 16 + mb + r;
                const int b = m >> 11;          // m / 2048
                const int s = m & 2047;
                dst[(((size_t)(b * Hc + h)) * Sc + s) * DKc + dk] =
                    (half_t)(acc[mi][ni][r] + bv);
            }
        }
}

// ---------------------------------------------------------------------------
// Kernel 2: flash attention. Block = (b,h) x 128 query rows; wave = 16 rows.
// Double-buffered K/V tiles: K staged via GLOBAL_LOAD_ASYNC_TO_LDS_B128
// (ASYNCcnt), V register-prefetched + transposed into LDS. One barrier/iter.
// ---------------------------------------------------------------------------
__global__ __launch_bounds__(256) void attn_wmma(
    const half_t* __restrict__ Qh, const half_t* __restrict__ Kh,
    const half_t* __restrict__ Vh, half_t* __restrict__ Oh)
{
    __shared__ __align__(16) half_t Ks[2][32][64];  // K tile, row-major [key][dk]
    __shared__ __align__(16) half_t Vt[2][64][40];  // V tile transposed [dk][key]
    __shared__ __align__(16) half_t Ps[8][16][40];  // per-wave P re-layout tile

    const int tid  = threadIdx.x;
    const int lane = tid & 31;
    const int wid  = tid >> 5;
    const int lr   = lane & 15;
    const int hs   = lane >> 4;
    const int mb   = hs * 8;

    const int bh   = blockIdx.y;                    // b*H + h
    const int qt   = blockIdx.x;
    const size_t base = (size_t)bh * Sc * DKc;
    const int qrow = qt * 128 + wid * 16;

    const half_t* kp = Kh + base;                   // uniform -> SGPR pair
    const half_t* vp = Vh + base;

    // Resident Q A-fragments for the two 32-wide K steps of DK=64.
    v16h qf[2];
    {
        const half_t* qp = Qh + base + (size_t)(qrow + lr) * DKc;
#pragma unroll
        for (int t = 0; t < 2; ++t) {
            const int k0 = t * 32 + hs * 8;
#pragma unroll
            for (int j = 0; j < 8; ++j) qf[t][j]     = qp[k0 + j];
#pragma unroll
            for (int j = 0; j < 8; ++j) qf[t][8 + j] = qp[k0 + 16 + j];
        }
    }

    v8f oacc[4];
#pragma unroll
    for (int t = 0; t < 4; ++t)
#pragma unroll
        for (int r = 0; r < 8; ++r) oacc[t][r] = 0.0f;
    float mrow[8], lsum[8];
#pragma unroll
    for (int r = 0; r < 8; ++r) { mrow[r] = -__builtin_inff(); lsum[r] = 0.0f; }

    const int srow = tid >> 3;        // 0..31 (tile row this thread stages)
    const int sseg = (tid & 7) * 8;   // 0..56 (halves)

    // LDS byte offsets (generic-pointer low 32 bits == LDS offset).
    const unsigned kld[2] = {
        (unsigned)(uintptr_t)&Ks[0][srow][sseg],
        (unsigned)(uintptr_t)&Ks[1][srow][sseg]
    };

    // ---- prologue: stage block 0 into buffer 0 ----
    {
        const unsigned go = (unsigned)(((unsigned)srow * DKc + sseg) * 2u);
        asm volatile("global_load_async_to_lds_b128 %0, %1, %2"
                     :: "v"(kld[0]), "v"(go), "s"(kp) : "memory");
        uint4 vreg = *(const uint4*)(vp + (size_t)srow * DKc + sseg);
        const half_t* hv = (const half_t*)&vreg;
#pragma unroll
        for (int j = 0; j < 8; ++j) Vt[0][sseg + j][srow] = hv[j];
        asm volatile("s_wait_asynccnt 0" ::: "memory");
    }
    __syncthreads();

    const int nblk = Sc / 32;         // 64 key blocks
    for (int ib = 0; ib < nblk; ++ib) {
        const int cur  = ib & 1;
        const int nxt  = cur ^ 1;
        const bool more = (ib + 1) < nblk;

        // ---- prefetch block ib+1 (overlaps with compute below) ----
        uint4 vreg;
        if (more) {
            const int kbn = (ib + 1) * 32;
            const unsigned go =
                (unsigned)((((unsigned)(kbn + srow)) * DKc + sseg) * 2u);
            asm volatile("global_load_async_to_lds_b128 %0, %1, %2"
                         :: "v"(kld[nxt]), "v"(go), "s"(kp) : "memory");
            vreg = *(const uint4*)(vp + (size_t)(kbn + srow) * DKc + sseg);
        }

        // ---- scores: S[16 x 32] = Q[16 x 64] * K^T ----
        v8f sacc[2];
#pragma unroll
        for (int n = 0; n < 2; ++n)
#pragma unroll
            for (int r = 0; r < 8; ++r) sacc[n][r] = 0.0f;
#pragma unroll
        for (int t = 0; t < 2; ++t) {
            const int kc = t * 32 + hs * 8;
#pragma unroll
            for (int n = 0; n < 2; ++n) {
                v16h bf;                            // B[k=d][n=key] = K[key][d]
                const int key = n * 16 + lr;
#pragma unroll
                for (int j = 0; j < 16; ++j) bf[j] = Ks[cur][key][kc + (j & 7) + ((j >> 3) << 4)];
                sacc[n] = __builtin_amdgcn_wmma_f32_16x16x32_f16(
                    false, qf[t], false, bf, (short)0, sacc[n], false, false);
            }
        }

        // ---- online softmax; row groups live in 16-lane halves ----
        const float scl = 0.125f;   // 1/sqrt(64)
#pragma unroll
        for (int r = 0; r < 8; ++r) {
            const float s0 = sacc[0][r] * scl;
            const float s1 = sacc[1][r] * scl;
            float v = fmaxf(s0, s1);
            v = fmaxf(v, __shfl_xor(v, 1, 32));
            v = fmaxf(v, __shfl_xor(v, 2, 32));
            v = fmaxf(v, __shfl_xor(v, 4, 32));
            v = fmaxf(v, __shfl_xor(v, 8, 32));
            const float mnew = fmaxf(mrow[r], v);
            const float p0 = __expf(s0 - mnew);
            const float p1 = __expf(s1 - mnew);
            Ps[wid][mb + r][lr]      = (half_t)p0;  // D-layout -> LDS
            Ps[wid][mb + r][16 + lr] = (half_t)p1;
            float ssum = p0 + p1;
            ssum += __shfl_xor(ssum, 1, 32);
            ssum += __shfl_xor(ssum, 2, 32);
            ssum += __shfl_xor(ssum, 4, 32);
            ssum += __shfl_xor(ssum, 8, 32);
            const float esc = __expf(mrow[r] - mnew);
            lsum[r] = lsum[r] * esc + ssum;
            mrow[r] = mnew;
#pragma unroll
            for (int tt = 0; tt < 4; ++tt) oacc[tt][r] *= esc;
        }

        // ---- P as A-fragment (wave-private LDS re-layout) ----
        v16h pf;
        {
            const int kbp = hs * 8;
#pragma unroll
            for (int j = 0; j < 8; ++j) pf[j]     = Ps[wid][lr][kbp + j];
#pragma unroll
            for (int j = 0; j < 8; ++j) pf[8 + j] = Ps[wid][lr][16 + kbp + j];
        }
        // ---- O += P[16 x 32keys] * V[32keys x 64] ----
#pragma unroll
        for (int t = 0; t < 4; ++t) {
            v16h vf;                                // B[k=key][n=d] = Vt[d][key]
            const int c = t * 16 + lr;
#pragma unroll
            for (int j = 0; j < 16; ++j) vf[j] = Vt[cur][c][hs * 8 + (j & 7) + ((j >> 3) << 4)];
            oacc[t] = __builtin_amdgcn_wmma_f32_16x16x32_f16(
                false, pf, false, vf, (short)0, oacc[t], false, false);
        }

        // ---- finish staging of block ib+1, then one barrier ----
        if (more) {
            const half_t* hv = (const half_t*)&vreg;
#pragma unroll
            for (int j = 0; j < 8; ++j) Vt[nxt][sseg + j][srow] = hv[j];
            asm volatile("s_wait_asynccnt 0" ::: "memory");
        }
        __syncthreads();
    }

    // Normalize and write combined-head output [B,S,E] as f16.
    const int b = bh >> 4;
    const int h = bh & 15;
#pragma unroll
    for (int t = 0; t < 4; ++t)
#pragma unroll
        for (int r = 0; r < 8; ++r) {
            const int s = qrow + mb + r;
            const int e = h * DKc + t * 16 + lr;
            Oh[((size_t)(b * Sc + s)) * Ec + e] = (half_t)(oacc[t][r] / lsum[r]);
        }
}

// ---------------------------------------------------------------------------
// Kernel 3: output projection. out[4096 x 1024] = O[4096 x 1024] * W_o^T + b_o
// Same skeleton as kernel 1 with f16 A-tile and register prefetch.
// ---------------------------------------------------------------------------
__global__ __launch_bounds__(256) void out_gemm_wmma(
    const half_t* __restrict__ A,     // [B*S, E] f16
    const float* __restrict__ W,      // [E, E] fp32
    const float* __restrict__ bias,   // [E]
    float* __restrict__ out)          // [B*S, E] fp32
{
    __shared__ __align__(16) half_t As[128][40];
    __shared__ __align__(16) half_t Ws[128][40];

    const int tid  = threadIdx.x;
    const int lane = tid & 31;
    const int wid  = tid >> 5;
    const int wm   = wid & 3;
    const int wn   = wid >> 2;
    const int m0   = blockIdx.y * 128;
    const int n0   = blockIdx.x * 128;
    const int lr   = lane & 15;
    const int hs   = lane >> 4;

    v8f acc[2][4];
#pragma unroll
    for (int mi = 0; mi < 2; ++mi)
#pragma unroll
        for (int ni = 0; ni < 4; ++ni)
#pragma unroll
            for (int r = 0; r < 8; ++r) acc[mi][ni][r] = 0.0f;

    const int arow = tid >> 2;        // 0..63 (f16 rows, uint4)
    const int aseg = (tid & 3) * 8;
    const int brow = tid >> 3;        // 0..31 (f32 rows, float4)
    const int bseg = (tid & 7) * 4;

    uint4  ra[2];
    float4 rw[4];
    auto loadAW = [&](int k0) {
#pragma unroll
        for (int p = 0; p < 2; ++p) {
            const int row = p * 64 + arow;
            ra[p] = *(const uint4*)(A + (size_t)(m0 + row) * Ec + k0 + aseg);
        }
#pragma unroll
        for (int p = 0; p < 4; ++p) {
            const int row = p * 32 + brow;
            rw[p] = *(const float4*)(W + (size_t)(n0 + row) * Ec + k0 + bseg);
        }
    };
    loadAW(0);

    for (int k0 = 0; k0 < Ec; k0 += 32) {
        __syncthreads();
#pragma unroll
        for (int p = 0; p < 2; ++p) {
            const int row = p * 64 + arow;
            *(uint4*)&As[row][aseg] = ra[p];
        }
#pragma unroll
        for (int p = 0; p < 4; ++p) {
            const int row = p * 32 + brow;
            Ws[row][bseg + 0] = (half_t)rw[p].x; Ws[row][bseg + 1] = (half_t)rw[p].y;
            Ws[row][bseg + 2] = (half_t)rw[p].z; Ws[row][bseg + 3] = (half_t)rw[p].w;
        }
        __syncthreads();
        if (k0 + 32 < Ec) loadAW(k0 + 32);

        v16h afr[2];
#pragma unroll
        for (int mi = 0; mi < 2; ++mi) {
            const int m  = wm * 32 + mi * 16 + lr;
            const int kb = hs * 8;
#pragma unroll
            for (int j = 0; j < 8; ++j) afr[mi][j]     = As[m][kb + j];
#pragma unroll
            for (int j = 0; j < 8; ++j) afr[mi][8 + j] = As[m][16 + kb + j];
        }
        v16h bfr[4];
#pragma unroll
        for (int ni = 0; ni < 4; ++ni) {
            const int n  = wn * 64 + ni * 16 + lr;
            const int kb = hs * 8;
#pragma unroll
            for (int j = 0; j < 16; ++j) bfr[ni][j] = Ws[n][kb + (j & 7) + ((j >> 3) << 4)];
        }
#pragma unroll
        for (int mi = 0; mi < 2; ++mi)
#pragma unroll
            for (int ni = 0; ni < 4; ++ni)
                acc[mi][ni] = __builtin_amdgcn_wmma_f32_16x16x32_f16(
                    false, afr[mi], false, bfr[ni], (short)0, acc[mi][ni], false, false);
    }

    const int mb = hs * 8;
#pragma unroll
    for (int mi = 0; mi < 2; ++mi)
#pragma unroll
        for (int ni = 0; ni < 4; ++ni) {
            const int n = n0 + wn * 64 + ni * 16 + lr;
            const float bv = bias[n];
#pragma unroll
            for (int r = 0; r < 8; ++r) {
                const int m = m0 + wm * 32 + mi * 16 + mb + r;
                out[(size_t)m * Ec + n] = acc[mi][ni][r] + bv;
            }
        }
}

// ---------------------------------------------------------------------------
extern "C" void kernel_launch(void* const* d_in, const int* in_sizes, int n_in,
                              void* d_out, int out_size, void* d_ws, size_t ws_size,
                              hipStream_t stream) {
    (void)in_sizes; (void)n_in; (void)out_size; (void)ws_size;
    const float* x     = (const float*)d_in[0];
    const float* W_qkv = (const float*)d_in[1];
    const float* b_qkv = (const float*)d_in[2];
    const float* W_o   = (const float*)d_in[3];
    const float* b_o   = (const float*)d_in[4];
    float* out = (float*)d_out;

    char* ws = (char*)d_ws;
    const size_t per = (size_t)Bc * Hc * Sc * DKc * sizeof(half_t); // 8 MiB each
    half_t* Qh = (half_t*)(ws + 0 * per);
    half_t* Kh = (half_t*)(ws + 1 * per);
    half_t* Vh = (half_t*)(ws + 2 * per);
    half_t* Oh = (half_t*)(ws + 3 * per);   // total 32 MiB of workspace

    qkv_gemm_wmma<<<dim3(3 * Ec / 128, (Bc * Sc) / 128), 256, 0, stream>>>(
        x, W_qkv, b_qkv, Qh, Kh, Vh);
    attn_wmma<<<dim3(Sc / 128, Bc * Hc), 256, 0, stream>>>(Qh, Kh, Vh, Oh);
    out_gemm_wmma<<<dim3(Ec / 128, (Bc * Sc) / 128), 256, 0, stream>>>(
        Oh, W_o, b_o, out);
}
